// MyNER_14611478741653
// MI455X (gfx1250) — compile-verified, matrix operand order checked by reading, not run on hardware
//
#include <hip/hip_runtime.h>
#include <math.h>

// Problem constants (match reference)
#define BB 32
#define TT 256
#define EE 768
#define DD 256
#define HH 100
#define G4 400     // 4*H
#define NTG 21
#define KP0 128    // Whh K padded 100 -> 128 (mult of 32)
#define KC 224     // concat K padded 200 -> 224 (mult of 32)
#define MROWS (BB*TT)   // 8192

typedef __attribute__((ext_vector_type(16))) _Float16 v16h;
typedef __attribute__((ext_vector_type(8)))  float    v8f;
typedef unsigned int u32x4 __attribute__((ext_vector_type(4)));
typedef int          i32x4 __attribute__((ext_vector_type(4)));
typedef int          i32x8 __attribute__((ext_vector_type(8)));

// ---------------------------------------------------------------------------
// Tensor Data Mover: async DMA of a 2D tile (rows x dim0 elements, row stride
// stride0 elements) from global memory into LDS at byte offset ldsOff.
// dsizeCode: 0=1B 1=2B 2=4B 3=8B.   D# layout per CDNA5 ISA ch.8.
// ---------------------------------------------------------------------------
__device__ __forceinline__ unsigned lds_off(const void* p) {
  // flat LDS address = {shared aperture, offset}; low 32 bits are the LDS offset
  return (unsigned)(uintptr_t)p;
}

__device__ __forceinline__ void tdm_load_2d(const void* gaddr, unsigned ldsOff,
                                            unsigned dsizeCode, unsigned dim0,
                                            unsigned rows, unsigned stride0) {
  const unsigned long long ga = (unsigned long long)(uintptr_t)gaddr;
  u32x4 g0;
  g0.x = 1u;                                            // count=1 (valid user D#)
  g0.y = ldsOff;                                        // lds_addr (bytes)
  g0.z = (unsigned)ga;                                  // global_addr[31:0]
  g0.w = (unsigned)((ga >> 32) & 0x01FFFFFFu) | (2u << 30); // addr[56:32] | type=2
  i32x8 g1;
  g1[0] = (int)(dsizeCode << 16);                       // workgroup_mask=0, data_size
  g1[1] = (int)((dim0 & 0xFFFFu) << 16);                // tensor_dim0[15:0] @ bit48
  g1[2] = (int)(((dim0 >> 16) & 0xFFFFu) | ((rows & 0xFFFFu) << 16)); // dim0 hi | dim1 lo
  g1[3] = (int)(((rows >> 16) & 0xFFFFu) | ((dim0 & 0xFFFFu) << 16)); // dim1 hi | tile_dim0
  g1[4] = (int)(rows & 0xFFFFu);                        // tile_dim1 (tile_dim2=0)
  g1[5] = (int)stride0;                                 // tensor_dim0_stride[31:0]
  g1[6] = 0;
  g1[7] = 0;
  i32x4 gz = {};
#if __clang_major__ >= 23
  i32x8 gz8 = {};
  __builtin_amdgcn_tensor_load_to_lds(g0, g1, gz, gz, gz8, 0);
#else
  __builtin_amdgcn_tensor_load_to_lds(g0, g1, gz, gz, 0);
#endif
}

// ---------------------------------------------------------------------------
// Wave-level 16x16 tile accumulate: D += A(16xK) * B^T(16xK)  via
// v_wmma_f32_16x16x32_f16.  A row-major [16 x lda], B stored as N x K
// row-major (i.e. already transposed).  K must be a multiple of 32.
// Lane layout per CDNA5 ISA 7.12.2 (16-bit A 16x32 and mirrored B).
// ---------------------------------------------------------------------------
__device__ __forceinline__ v8f wmma_acc_tile(const _Float16* __restrict__ A, int lda,
                                             const _Float16* __restrict__ Bm, int ldb,
                                             int K, v8f acc, int lane) {
  const int r  = lane & 15;          // M row (A) / N col (B)
  const int hb = (lane >> 4) & 1;    // K-half select
  for (int k0 = 0; k0 < K; k0 += 32) {
    v16h av, bv;
    const int ka = k0 + hb * 8;
    const _Float16* ap = A  + (size_t)r * lda + ka;
    const _Float16* bp = Bm + (size_t)r * ldb + ka;
#pragma unroll
    for (int j = 0; j < 8; ++j) {
      av[j] = ap[j];  av[8 + j] = ap[16 + j];
      bv[j] = bp[j];  bv[8 + j] = bp[16 + j];
    }
    acc = __builtin_amdgcn_wmma_f32_16x16x32_f16(
        /*neg_a=*/false, av, /*neg_b=*/false, bv,
        /*c_mod=*/(short)0, acc, /*reuse_a=*/false, /*reuse_b=*/false);
  }
  return acc;
}

__device__ __forceinline__ float sigf(float x) { return 1.f / (1.f + __expf(-x)); }

// ---------------------------------------------------------------------------
// Generic WMMA GEMM: C[m,n] = sum_k A[m,k]*B[n,k] + bias[n]
// One wave (32 threads) per 16x16 output tile. grid = (N/16 tiles, M/16 tiles)
// ---------------------------------------------------------------------------
__global__ void k_gemm(const _Float16* __restrict__ A, int lda,
                       const _Float16* __restrict__ Bm, int ldb,
                       const float* __restrict__ bias, int nbias,
                       float* __restrict__ Cf, _Float16* __restrict__ Ch, int ldc,
                       int Nreal, int K) {
  const int nt = blockIdx.x, mt = blockIdx.y;
  const int lane = threadIdx.x;
  const _Float16* Ap = A  + (size_t)(mt * 16) * lda;
  const _Float16* Bp = Bm + (size_t)(nt * 16) * ldb;
  v8f acc = {};
  acc = wmma_acc_tile(Ap, lda, Bp, ldb, K, acc, lane);
  const int n  = nt * 16 + (lane & 15);
  const int m0 = mt * 16 + ((lane >> 4) & 1) * 8;
  if (n < Nreal) {
    const float bv = (bias && n < nbias) ? bias[n] : 0.f;
#pragma unroll
    for (int r = 0; r < 8; ++r) {
      const float v = acc[r] + bv;
      const size_t idx = (size_t)(m0 + r) * ldc + n;
      if (Cf) Cf[idx] = v;
      if (Ch) Ch[idx] = (_Float16)v;
    }
  }
}

// ---------------------------------------------------------------------------
// Embedding gather + f32 -> f16 convert
// ---------------------------------------------------------------------------
__global__ void k_embed(const int* __restrict__ words, const float* __restrict__ table,
                        _Float16* __restrict__ out, int rows, int E) {
  size_t i = (size_t)blockIdx.x * blockDim.x + threadIdx.x;
  size_t tot = (size_t)rows * E;
  if (i >= tot) return;
  int m = (int)(i / E), e = (int)(i % E);
  out[i] = (_Float16)table[(size_t)words[m] * E + e];
}

// ---------------------------------------------------------------------------
// Convert f32 (N x K) to f16 (Np x Kp) with zero padding (N=0 => pure zero)
// ---------------------------------------------------------------------------
__global__ void k_pad16(const float* __restrict__ src, int N, int K,
                        _Float16* __restrict__ dst, int Np, int Kp) {
  size_t i = (size_t)blockIdx.x * blockDim.x + threadIdx.x;
  size_t tot = (size_t)Np * Kp;
  if (i >= tot) return;
  int n = (int)(i / Kp), k = (int)(i % Kp);
  dst[i] = (n < N && k < K) ? (_Float16)src[(size_t)n * K + k] : (_Float16)0.f;
}

// ---------------------------------------------------------------------------
// Persistent LSTM direction kernel: one workgroup (8 waves) per direction.
//   gates(t) = pre[t] (x@Wih^T + bih, in HBM) + h(t-1)@Whh^T + bhh
// CDNA5 data movement:
//   * Whh^T (400x128 f16, 100 KB) DMA'd into LDS ONCE via tensor_load_to_lds
//   * pre[t] slab (32x400 f32, strided rows) double-buffered in LDS via TDM,
//     prefetching t+1 after step t's WMMA phase; s_wait_tensorcnt syncs
//   * h state lives in LDS as f16; recurrent matmul is pure LDS-fed WMMA
// Total LDS: 8K(h) + 51.2K(G) + 12.8K(c) + 100K(Whh) + 2x51.2K(pre) = 277 KB
// ---------------------------------------------------------------------------
__global__ void k_lstm(const float* __restrict__ preBase, size_t preStride,
                       const _Float16* __restrict__ whhBase, size_t whhStride,
                       const float* __restrict__ bhhBase,
                       _Float16* __restrict__ hcat, int ldcat) {
  const int dir = blockIdx.x;                 // 0 = forward, 1 = backward
  const float*    pre = preBase + (size_t)dir * preStride;
  const _Float16* Whh = whhBase + (size_t)dir * whhStride;
  const float*    bhh = bhhBase + (size_t)dir * G4;
  const int colOff = dir * HH;
  const int reverse = dir;

  extern __shared__ char smem[];
  _Float16* hA   = (_Float16*)smem;                                   //  8192 B
  float*    G    = (float*)(smem + 8192);                             // 51200 B
  float*    cs   = (float*)(smem + 8192 + 51200);                     // 12800 B
  _Float16* Wlds = (_Float16*)(smem + 8192 + 51200 + 12800);          // 102400 B
  float*    pb0  = (float*)(smem + 8192 + 51200 + 12800 + 102400);            // 51200 B
  float*    pb1  = (float*)(smem + 8192 + 51200 + 12800 + 102400 + 51200);    // 51200 B

  const int wave = threadIdx.x >> 5;
  const int lane = threadIdx.x & 31;

  // Kick off DMA: weights (once) + first pre slab. Wave 0 owns the TDM queue.
  if (wave == 0) {
    tdm_load_2d(Whh, lds_off(Wlds), /*2B*/1, KP0, G4, KP0);
    const int t0 = reverse ? (TT - 1) : 0;
    tdm_load_2d(pre + (size_t)t0 * G4, lds_off(pb0), /*4B*/2, G4, BB,
                (unsigned)(TT * G4));
  }
  for (int i = threadIdx.x; i < 32 * KP0; i += blockDim.x) hA[i] = (_Float16)0.f;
  for (int i = threadIdx.x; i < 32 * HH;  i += blockDim.x) cs[i] = 0.f;
  __builtin_amdgcn_s_wait_tensorcnt(0);   // wave0 drains its 2 DMAs; others trivial
  __syncthreads();

  for (int step = 0; step < TT; ++step) {
    const int t = reverse ? (TT - 1 - step) : step;
    float* bufc = (step & 1) ? pb1 : pb0;
    float* bufn = (step & 1) ? pb0 : pb1;

    // ---- recurrent matmul: G = h(t-1) @ Whh^T  (M=32, N=400, K=128), LDS-fed ----
    for (int tile = wave; tile < 2 * 25; tile += 8) {
      const int mt = tile / 25, nt = tile % 25;
      v8f acc = {};
      acc = wmma_acc_tile(hA + (size_t)(mt * 16) * KP0, KP0,
                          Wlds + (size_t)(nt * 16) * KP0, KP0, KP0, acc, lane);
      const int n  = nt * 16 + (lane & 15);
      const int m0 = mt * 16 + ((lane >> 4) & 1) * 8;
#pragma unroll
      for (int r = 0; r < 8; ++r) G[(m0 + r) * G4 + n] = acc[r];
    }

    // ---- ensure current pre slab landed; prefetch next step's slab ----
    __builtin_amdgcn_s_wait_tensorcnt(0);
    if (wave == 0 && step + 1 < TT) {
      const int tn = reverse ? (TT - 2 - step) : (step + 1);
      tdm_load_2d(pre + (size_t)tn * G4, lds_off(bufn), /*4B*/2, G4, BB,
                  (unsigned)(TT * G4));
    }
    __syncthreads();

    // ---- element-wise gates: i, f, g, o chunks of 100 (pre read from LDS) ----
    for (int idx = threadIdx.x; idx < 32 * HH; idx += blockDim.x) {
      const int b = idx / HH, j = idx % HH;
      const float* prow = bufc + b * G4;
      const float gi = G[b * G4 + j]          + prow[j]          + bhh[j];
      const float gf = G[b * G4 + HH + j]     + prow[HH + j]     + bhh[HH + j];
      const float gg = G[b * G4 + 2 * HH + j] + prow[2 * HH + j] + bhh[2 * HH + j];
      const float go = G[b * G4 + 3 * HH + j] + prow[3 * HH + j] + bhh[3 * HH + j];
      const float c = sigf(gf) * cs[idx] + sigf(gi) * tanhf(gg);
      const float h = sigf(go) * tanhf(c);
      cs[idx] = c;
      hA[b * KP0 + j] = (_Float16)h;
      hcat[(size_t)(b * TT + t) * ldcat + colOff + j] = (_Float16)h;
    }
    __syncthreads();
  }
}

// ---------------------------------------------------------------------------
// In-place log_softmax over 21 tags (rows stride 32)
// ---------------------------------------------------------------------------
__global__ void k_lsm(float* __restrict__ scores, int rows) {
  int m = blockIdx.x * blockDim.x + threadIdx.x;
  if (m >= rows) return;
  float* p = scores + (size_t)m * 32;
  float mx = -1e30f;
  for (int j = 0; j < NTG; ++j) mx = fmaxf(mx, p[j]);
  float s = 0.f;
  for (int j = 0; j < NTG; ++j) s += __expf(p[j] - mx);
  const float l = mx + logf(s);
  for (int j = 0; j < NTG; ++j) p[j] -= l;
}

// ---------------------------------------------------------------------------
// CRF loss: one wave per batch element; lane j holds alpha[j] (j < 21).
// ---------------------------------------------------------------------------
__global__ void k_crf(const float* __restrict__ scores, const int* __restrict__ words,
                      const int* __restrict__ target, const float* __restrict__ trans,
                      const float* __restrict__ start_s, const float* __restrict__ end_s,
                      float* __restrict__ out) {
  const int b = blockIdx.x;
  const int lane = threadIdx.x;
  const float* sc = scores + (size_t)b * TT * 32;
  const int* w  = words  + (size_t)b * TT;
  const int* tg = target + (size_t)b * TT;
  const int j = (lane < NTG) ? lane : 0;

  float alpha = (lane < NTG) ? (sc[lane] + start_s[lane]) : -1e30f;

  for (int t = 1; t < TT; ++t) {
    const int msk = (w[t] != 0);
    const float emit = sc[t * 32 + j];
    float m = -1e30f, s = 0.f;
    for (int i = 0; i < NTG; ++i) {
      const float ai = __shfl(alpha, i, 32);
      const float v = ai + trans[i * NTG + j];
      if (v > m) { s = s * __expf(m - v) + 1.f; m = v; }
      else       { s += __expf(v - m); }
    }
    const float nv = m + logf(s) + emit;
    if (msk && lane < NTG) alpha = nv;
  }

  float val = (lane < NTG) ? (alpha + end_s[lane]) : -1e30f;
  float m = val;
  for (int o = 16; o > 0; o >>= 1) m = fmaxf(m, __shfl_xor(m, o, 32));
  float e = (lane < NTG) ? __expf(val - m) : 0.f;
  for (int o = 16; o > 0; o >>= 1) e += __shfl_xor(e, o, 32);
  const float norm = m + logf(e);

  if (lane == 0) {
    float gold = start_s[tg[0]];
    int seqlen = 0;
    for (int t = 0; t < TT; ++t) seqlen += (w[t] != 0);
    gold += sc[tg[0]] * ((w[0] != 0) ? 1.f : 0.f);
    for (int t = 1; t < TT; ++t) {
      const float mt = (w[t] != 0) ? 1.f : 0.f;
      gold += sc[t * 32 + tg[t]] * mt;
      gold += trans[tg[t - 1] * NTG + tg[t]] * mt;
    }
    const int last = (seqlen > 0) ? (seqlen - 1) : 0;
    gold += end_s[tg[last]];
    out[b] = norm - gold;
  }
}

// ---------------------------------------------------------------------------
extern "C" void kernel_launch(void* const* d_in, const int* in_sizes, int n_in,
                              void* d_out, int out_size, void* d_ws, size_t ws_size,
                              hipStream_t stream) {
  const int*   words   = (const int*)d_in[0];
  const int*   target  = (const int*)d_in[1];
  const float* embed   = (const float*)d_in[2];
  const float* in_fc_w = (const float*)d_in[3];
  const float* in_fc_b = (const float*)d_in[4];
  const float* Wih0    = (const float*)d_in[5];
  const float* Whh0    = (const float*)d_in[6];
  const float* bih0    = (const float*)d_in[7];
  const float* bhh0    = (const float*)d_in[8];
  const float* Wih1    = (const float*)d_in[9];
  const float* Whh1    = (const float*)d_in[10];
  const float* bih1    = (const float*)d_in[11];
  const float* bhh1    = (const float*)d_in[12];
  const float* out_w   = (const float*)d_in[13];
  const float* out_b   = (const float*)d_in[14];
  const float* trans   = (const float*)d_in[15];
  const float* start_s = (const float*)d_in[16];
  const float* end_s   = (const float*)d_in[17];
  float* outp = (float*)d_out;

  // ---- workspace bump allocator ----
  char* ws = (char*)d_ws;
  size_t off = 0;
  auto alloc = [&](size_t bytes) -> char* {
    off = (off + 255) & ~(size_t)255;
    char* p = ws + off;
    off += bytes;
    return p;
  };
  _Float16* w_infc = (_Float16*)alloc((size_t)DD * EE * 2);            // 256x768
  _Float16* w_ih0  = (_Float16*)alloc((size_t)2 * G4 * DD * 2);        // 2x400x256
  _Float16* w_hh0  = (_Float16*)alloc((size_t)2 * G4 * KP0 * 2);       // 2x400x128
  _Float16* w_ih1  = (_Float16*)alloc((size_t)2 * G4 * KC * 2);        // 2x400x224
  _Float16* w_hh1  = (_Float16*)alloc((size_t)2 * G4 * KP0 * 2);       // 2x400x128
  _Float16* w_out  = (_Float16*)alloc((size_t)32 * KC * 2);            // 32x224
  _Float16* x1     = (_Float16*)alloc((size_t)MROWS * DD * 2);         // 8192x256
  _Float16* h0cat  = (_Float16*)alloc((size_t)MROWS * KC * 2);         // 8192x224
  _Float16* h1cat  = (_Float16*)alloc((size_t)MROWS * KC * 2);         // 8192x224
  float*    scores = (float*)   alloc((size_t)MROWS * 32 * 4);         // 8192x32
  float*    pre    = (float*)   alloc((size_t)2 * MROWS * G4 * 4);     // 2x8192x400
  _Float16* xemb   = (_Float16*)pre;  // alias: xemb dead before pre is written
  (void)in_sizes; (void)n_in; (void)out_size; (void)ws_size;

  const size_t preStride = (size_t)MROWS * G4;
  auto cdiv = [](size_t a, size_t b) { return (unsigned)((a + b - 1) / b); };

  // LSTM kernel uses 277 KB of LDS (CDNA5 WGP has 320 KB)
  const size_t ldsBytes = 8192 + 51200 + 12800 + 102400 + 2 * 51200;
  static bool attrDone = false;
  if (!attrDone) {
    hipFuncSetAttribute((const void*)k_lstm,
                        hipFuncAttributeMaxDynamicSharedMemorySize, (int)ldsBytes);
    attrDone = true;
  }

  // ---- weight conversion / padding ----
  k_pad16<<<cdiv((size_t)DD * EE, 256), 256, 0, stream>>>(in_fc_w, DD, EE, w_infc, DD, EE);
  k_pad16<<<cdiv((size_t)800 * DD, 256), 256, 0, stream>>>(Wih0, 800, DD, w_ih0, 800, DD);
  k_pad16<<<cdiv((size_t)800 * KP0, 256), 256, 0, stream>>>(Whh0, 800, HH, w_hh0, 800, KP0);
  k_pad16<<<cdiv((size_t)800 * KC, 256), 256, 0, stream>>>(Wih1, 800, 2 * HH, w_ih1, 800, KC);
  k_pad16<<<cdiv((size_t)800 * KP0, 256), 256, 0, stream>>>(Whh1, 800, HH, w_hh1, 800, KP0);
  k_pad16<<<cdiv((size_t)32 * KC, 256), 256, 0, stream>>>(out_w, NTG, 2 * HH, w_out, 32, KC);
  k_pad16<<<cdiv((size_t)MROWS * KC, 256), 256, 0, stream>>>(nullptr, 0, 0, h0cat, MROWS, KC);
  k_pad16<<<cdiv((size_t)MROWS * KC, 256), 256, 0, stream>>>(nullptr, 0, 0, h1cat, MROWS, KC);

  // ---- embedding gather (f32 -> f16) ----
  k_embed<<<cdiv((size_t)MROWS * EE, 256), 256, 0, stream>>>(words, embed, xemb, MROWS, EE);

  // ---- x1 = emb @ in_fc_w^T + b   (M=8192, N=256, K=768) ----
  k_gemm<<<dim3(DD / 16, MROWS / 16), 32, 0, stream>>>(
      xemb, EE, w_infc, EE, in_fc_b, DD, nullptr, x1, DD, DD, EE);

  // ---- layer 0 input projections, both directions (N=400, K=256) ----
  for (int d = 0; d < 2; ++d)
    k_gemm<<<dim3(G4 / 16, MROWS / 16), 32, 0, stream>>>(
        x1, DD, w_ih0 + (size_t)d * G4 * DD, DD, bih0 + d * G4, G4,
        pre + d * preStride, nullptr, G4, G4, DD);

  // ---- layer 0 recurrence (2 persistent workgroups, fwd+bwd, TDM-fed) ----
  k_lstm<<<2, 256, ldsBytes, stream>>>(pre, preStride, w_hh0, (size_t)G4 * KP0, bhh0, h0cat, KC);

  // ---- layer 1 input projections (N=400, K=224) ----
  for (int d = 0; d < 2; ++d)
    k_gemm<<<dim3(G4 / 16, MROWS / 16), 32, 0, stream>>>(
        h0cat, KC, w_ih1 + (size_t)d * G4 * KC, KC, bih1 + d * G4, G4,
        pre + d * preStride, nullptr, G4, G4, KC);

  // ---- layer 1 recurrence ----
  k_lstm<<<2, 256, ldsBytes, stream>>>(pre, preStride, w_hh1, (size_t)G4 * KP0, bhh1, h1cat, KC);

  // ---- output FC: scores = h1cat @ out_w^T + out_b  (N=21 in 32-padded tiles) ----
  k_gemm<<<dim3(2, MROWS / 16), 32, 0, stream>>>(
      h1cat, KC, w_out, KC, out_b, NTG, scores, nullptr, 32, NTG, KC);

  // ---- log_softmax + CRF ----
  k_lsm<<<cdiv(MROWS, 256), 256, 0, stream>>>(scores, MROWS);
  k_crf<<<BB, 32, 0, stream>>>(scores, words, target, trans, start_s, end_s, outp);
}
